// TreeAttention_65687229825622
// MI455X (gfx1250) — matrix operand order repaired and checked by reading, not compiled, not run on hardware
//
#include <hip/hip_runtime.h>

typedef __attribute__((ext_vector_type(16))) _Float16 v16h;
typedef __attribute__((ext_vector_type(8)))  float    v8f;
typedef __attribute__((__vector_size__(16))) int      v4i_t;

#define T_SEQ  2048
#define DMODEL 2048
#define NHEADS 16
#define HDIM   128

#if __has_builtin(__builtin_amdgcn_global_load_async_to_lds_b128)
#define HAVE_ASYNC_LDS 1
#endif

// POD vector types (avoid HIP class-type vectors inside unions)
struct __attribute__((aligned(16))) U4 { unsigned int x, y, z, w; };
struct __attribute__((aligned(8)))  U2 { unsigned int x, y; };
struct __attribute__((aligned(16))) F4 { float x, y, z, w; };

union Frag  { U4 u[2]; v16h h; };           // 32B = one 16-half WMMA fragment
union Pack4 { _Float16 h[4]; U2 u; };       // 4 halves <-> 8 bytes

static __device__ inline v8f zero8() {
  v8f z;
#pragma unroll
  for (int i = 0; i < 8; ++i) z[i] = 0.0f;
  return z;
}

static __device__ inline v16h ld_frag2(const _Float16* p0, const _Float16* p1) {
  Frag f;
  f.u[0] = *(const U4*)p0;
  f.u[1] = *(const U4*)p1;
  return f.h;
}

static __device__ inline v8f wmma_f16(v16h a, v16h b, v8f c) {
  return __builtin_amdgcn_wmma_f32_16x16x32_f16(false, a, false, b, (short)0, c,
                                                false, false);
}

// ---------------------------------------------------------------------------
// GEMM: C[r][c] = sum_k A[r][k] * W[c][k]   (i.e. A @ W^T), 2048^3
// A16:   A is _Float16 (workspace) vs float (input x)
// OUTF32: write float to d_out vs _Float16 to workspace
// Block: 256x128 tile, BK=64, 256 threads = 8 waves (4x2), wave tile 64x64.
// Per 32-K step: 8 fragment loads (16 ds_load_b128) feed 16 WMMAs.
// ---------------------------------------------------------------------------
template <bool A16, bool OUTF32>
__global__ __launch_bounds__(256) void gemm_nt_kernel(const void* __restrict__ Ap,
                                                      const float* __restrict__ W,
                                                      void* __restrict__ Cp) {
  __shared__ alignas(16) _Float16 sA[256][72];   // 64 K + 8 pad (bank spread)
  __shared__ alignas(16) _Float16 sB[128][72];

  const int tid  = threadIdx.x;
  const int lane = tid & 31;
  const int wv   = tid >> 5;          // 0..7
  const int hs   = lane >> 4;         // lane half (0/1)
  const int ln   = lane & 15;
  const int wm   = wv >> 1;           // 0..3 -> 64-row group
  const int wn   = wv & 1;            // 0..1 -> 64-col group
  const int bm   = blockIdx.x * 256;
  const int bn   = blockIdx.y * 128;

  v8f acc[4][4];
#pragma unroll
  for (int mt = 0; mt < 4; ++mt)
#pragma unroll
    for (int nt = 0; nt < 4; ++nt) acc[mt][nt] = zero8();

  for (int k0 = 0; k0 < DMODEL; k0 += 64) {
    // ---- stage A tile (256 x 64) into LDS as f16 ----
    if (!A16) {
      const float* A = (const float*)Ap;
#pragma unroll
      for (int it = 0; it < 16; ++it) {
        int idx = tid + it * 256;          // float4 units: 256 rows * 16
        int r   = idx >> 4;
        int c4  = idx & 15;
        F4 f = *(const F4*)(A + (size_t)(bm + r) * DMODEL + k0 + c4 * 4);
        Pack4 p;
        p.h[0] = (_Float16)f.x; p.h[1] = (_Float16)f.y;
        p.h[2] = (_Float16)f.z; p.h[3] = (_Float16)f.w;
        *(U2*)&sA[r][c4 * 4] = p.u;
      }
    } else {
      const _Float16* A = (const _Float16*)Ap;
#pragma unroll
      for (int it = 0; it < 8; ++it) {
        int idx = tid + it * 256;          // 8-half units: 256 rows * 8
        int r   = idx >> 3;
        int c8  = idx & 7;
        *(U4*)&sA[r][c8 * 8] =
            *(const U4*)(A + (size_t)(bm + r) * DMODEL + k0 + c8 * 8);
      }
    }
    // ---- stage W tile (rows are output cols) ----
    {
#pragma unroll
      for (int it = 0; it < 8; ++it) {
        int idx = tid + it * 256;          // float4 units: 128 rows * 16
        int r   = idx >> 4;
        int c4  = idx & 15;
        F4 f = *(const F4*)(W + (size_t)(bn + r) * DMODEL + k0 + c4 * 4);
        Pack4 p;
        p.h[0] = (_Float16)f.x; p.h[1] = (_Float16)f.y;
        p.h[2] = (_Float16)f.z; p.h[3] = (_Float16)f.w;
        *(U2*)&sB[r][c4 * 4] = p.u;
      }
    }
    __syncthreads();

#pragma unroll
    for (int kk = 0; kk < 64; kk += 32) {
      v16h af[4], bf[4];
#pragma unroll
      for (int mt = 0; mt < 4; ++mt) {
        int r = wm * 64 + mt * 16 + ln;    // A: lane = M row; K chunks by half
        af[mt] = ld_frag2(&sA[r][kk + hs * 8], &sA[r][kk + 16 + hs * 8]);
      }
#pragma unroll
      for (int nt = 0; nt < 4; ++nt) {
        int r = wn * 64 + nt * 16 + ln;    // B: lane = N col; K chunks by half
        const _Float16* p = &sB[r][kk + hs * 8];
        bf[nt] = ld_frag2(p, p + 16);
      }
#pragma unroll
      for (int mt = 0; mt < 4; ++mt)
#pragma unroll
        for (int nt = 0; nt < 4; ++nt)
          acc[mt][nt] = wmma_f16(af[mt], bf[nt], acc[mt][nt]);
    }
    __syncthreads();
  }

#pragma unroll
  for (int mt = 0; mt < 4; ++mt)
#pragma unroll
    for (int nt = 0; nt < 4; ++nt)
#pragma unroll
      for (int i = 0; i < 8; ++i) {
        int gr = bm + wm * 64 + mt * 16 + i + hs * 8;
        int gc = bn + wn * 64 + nt * 16 + ln;
        if (OUTF32)
          ((float*)Cp)[(size_t)gr * DMODEL + gc] = acc[mt][nt][i];
        else
          ((_Float16*)Cp)[(size_t)gr * DMODEL + gc] = (_Float16)acc[mt][nt][i];
      }
}

// ---------------------------------------------------------------------------
// Flash attention: grid (16 q-tiles, 16 heads), 256 threads = 8 waves.
// Wave w owns query rows qtile*128 + w*16 .. +15; key blocks of 32.
// ---------------------------------------------------------------------------
__global__ __launch_bounds__(256) void attn_kernel(const _Float16* __restrict__ Q,
                                                   const _Float16* __restrict__ K,
                                                   const _Float16* __restrict__ V,
                                                   _Float16* __restrict__ O) {
  __shared__ alignas(16) _Float16 ksh[32][136];    // K rows [key][dim]
  __shared__ alignas(16) _Float16 vsh[128][40];    // V transposed [dim][key]
  __shared__ alignas(16) _Float16 psh[8][16][40];  // per-wave P scratch

  const int tid  = threadIdx.x;
  const int lane = tid & 31;
  const int wv   = tid >> 5;
  const int hs   = lane >> 4;
  const int ln   = lane & 15;
  const int qtile = blockIdx.x;
  const int head  = blockIdx.y;
  const int qrow0 = qtile * 128 + wv * 16;
  const size_t hoff = (size_t)head * HDIM;

  // Preload Q fragments (A-matrix, 4 K-steps over head dim 128)
  v16h qf[4];
  {
    const _Float16* qp = Q + (size_t)(qrow0 + ln) * DMODEL + hoff;
#pragma unroll
    for (int ks = 0; ks < 4; ++ks)
      qf[ks] = ld_frag2(qp + ks * 32 + hs * 8, qp + ks * 32 + 16 + hs * 8);
  }

  v8f oacc[8];
#pragma unroll
  for (int nt = 0; nt < 8; ++nt) oacc[nt] = zero8();
  float mrow[8], lrow[8];
#pragma unroll
  for (int i = 0; i < 8; ++i) { mrow[i] = -3.0e38f; lrow[i] = 0.0f; }

  const float rsc = 0.08838834764831845f;  // 1/sqrt(128)
  const int nkb = (qtile + 1) * 4;         // uniform over block -> barriers safe

  for (int kb = 0; kb < nkb; ++kb) {
    // stage K block (32 keys x 128 dims), row-major; async direct-to-LDS when
    // available (gfx1250 GLOBAL_LOAD_ASYNC_TO_LDS_B128, tracked by ASYNCcnt)
    {
      int r  = tid >> 3;       // key 0..31
      int c8 = tid & 7;        // 16-dim chunk
      const _Float16* kp = K + (size_t)(kb * 32 + r) * DMODEL + hoff + c8 * 16;
#if defined(HAVE_ASYNC_LDS)
      __builtin_amdgcn_global_load_async_to_lds_b128(
          (v4i_t*)kp, (v4i_t*)&ksh[r][c8 * 16], 0, 0);
      __builtin_amdgcn_global_load_async_to_lds_b128(
          (v4i_t*)(kp + 8), (v4i_t*)&ksh[r][c8 * 16 + 8], 0, 0);
#else
      *(U4*)&ksh[r][c8 * 16]     = *(const U4*)kp;
      *(U4*)&ksh[r][c8 * 16 + 8] = *(const U4*)(kp + 8);
#endif
    }
    // stage V block transposed -> vsh[dim][key] (transpose: keep manual copy)
    {
      int key = tid >> 3;
      int c8  = tid & 7;
      const _Float16* vp = V + (size_t)(kb * 32 + key) * DMODEL + hoff + c8 * 16;
      Frag f;
      f.u[0] = *(const U4*)vp;
      f.u[1] = *(const U4*)(vp + 8);
#pragma unroll
      for (int j = 0; j < 16; ++j) vsh[c8 * 16 + j][key] = f.h[j];
    }
#if defined(HAVE_ASYNC_LDS)
#if __has_builtin(__builtin_amdgcn_s_wait_asynccnt)
    __builtin_amdgcn_s_wait_asynccnt(0);
#else
    __asm__ volatile("s_wait_asynccnt 0" ::: "memory");
#endif
#endif
    __syncthreads();

    // S = Q K^T over this wave's 16 queries x 32 keys
    v8f s0 = zero8(), s1 = zero8();
#pragma unroll
    for (int ks = 0; ks < 4; ++ks) {
      const _Float16* k0p = &ksh[ln][ks * 32 + hs * 8];
      const _Float16* k1p = &ksh[16 + ln][ks * 32 + hs * 8];
      s0 = wmma_f16(qf[ks], ld_frag2(k0p, k0p + 16), s0);
      s1 = wmma_f16(qf[ks], ld_frag2(k1p, k1p + 16), s1);
    }

    // online softmax (rows split by lane half: lanes 0-15 -> i, 16-31 -> i+8)
    float p0[8], p1[8], al[8];
#pragma unroll
    for (int i = 0; i < 8; ++i) {
      int qg  = qrow0 + i + hs * 8;
      int kg0 = kb * 32 + ln;
      int kg1 = kg0 + 16;
      float a = s0[i] * rsc; if (kg0 > qg) a = -1.0e30f;
      float b = s1[i] * rsc; if (kg1 > qg) b = -1.0e30f;
      float rm = fmaxf(a, b);
      rm = fmaxf(rm, __shfl_xor(rm, 1, 16));
      rm = fmaxf(rm, __shfl_xor(rm, 2, 16));
      rm = fmaxf(rm, __shfl_xor(rm, 4, 16));
      rm = fmaxf(rm, __shfl_xor(rm, 8, 16));
      float mn    = fmaxf(mrow[i], rm);
      float alpha = __expf(mrow[i] - mn);
      float e0 = __expf(a - mn);
      float e1 = __expf(b - mn);
      float rs = e0 + e1;
      rs += __shfl_xor(rs, 1, 16);
      rs += __shfl_xor(rs, 2, 16);
      rs += __shfl_xor(rs, 4, 16);
      rs += __shfl_xor(rs, 8, 16);
      lrow[i] = lrow[i] * alpha + rs;
      mrow[i] = mn;
      p0[i] = e0; p1[i] = e1; al[i] = alpha;
    }

    // rescale running O
#pragma unroll
    for (int nt = 0; nt < 8; ++nt)
#pragma unroll
      for (int i = 0; i < 8; ++i) oacc[nt][i] *= al[i];

    // C-layout P -> per-wave LDS -> A-layout fragment (intra-wave, in-order)
#pragma unroll
    for (int i = 0; i < 8; ++i) {
      int r = i + hs * 8;
      psh[wv][r][ln]      = (_Float16)p0[i];
      psh[wv][r][16 + ln] = (_Float16)p1[i];
    }
    __asm__ volatile("s_wait_dscnt 0" ::: "memory");

    v16h pf;
    {
      const _Float16* pp = &psh[wv][ln][0];
      pf = ld_frag2(pp + hs * 8, pp + 16 + hs * 8);
    }

    // O += P @ V  (B fragment: key chunks by half from transposed V)
#pragma unroll
    for (int nt = 0; nt < 8; ++nt) {
      const _Float16* vp = &vsh[nt * 16 + ln][hs * 8];
      oacc[nt] = wmma_f16(pf, ld_frag2(vp, vp + 16), oacc[nt]);
    }
    __syncthreads();
  }

  // normalize and store f16 attention output [T][DMODEL]
  float inv[8];
#pragma unroll
  for (int i = 0; i < 8; ++i) inv[i] = 1.0f / lrow[i];
#pragma unroll
  for (int nt = 0; nt < 8; ++nt)
#pragma unroll
    for (int i = 0; i < 8; ++i) {
      size_t r = (size_t)(qrow0 + i + hs * 8);
      O[r * DMODEL + hoff + nt * 16 + ln] = (_Float16)(oacc[nt][i] * inv[i]);
    }
}

// ---------------------------------------------------------------------------
extern "C" void kernel_launch(void* const* d_in, const int* in_sizes, int n_in,
                              void* d_out, int out_size, void* d_ws, size_t ws_size,
                              hipStream_t stream) {
  const float* x  = (const float*)d_in[0];
  const float* Wq = (const float*)d_in[1];
  const float* Wk = (const float*)d_in[2];
  const float* Wv = (const float*)d_in[3];
  const float* Wo = (const float*)d_in[4];

  const size_t TEN = (size_t)T_SEQ * DMODEL;  // 4M elements
  _Float16* Q16 = (_Float16*)d_ws;
  _Float16* K16 = Q16 + TEN;
  _Float16* V16 = K16 + TEN;
  _Float16* O16 = V16 + TEN;                  // total 32 MB f16 workspace

  dim3 gg(DMODEL / 256, DMODEL / 128);        // (8,16)
  gemm_nt_kernel<false, false><<<gg, 256, 0, stream>>>((const void*)x, Wq, (void*)Q16);
  gemm_nt_kernel<false, false><<<gg, 256, 0, stream>>>((const void*)x, Wk, (void*)K16);
  gemm_nt_kernel<false, false><<<gg, 256, 0, stream>>>((const void*)x, Wv, (void*)V16);

  dim3 ga(T_SEQ / 128, NHEADS);               // (16,16)
  attn_kernel<<<ga, 256, 0, stream>>>(Q16, K16, V16, O16);

  gemm_nt_kernel<true, true><<<gg, 256, 0, stream>>>((const void*)O16, Wo, d_out);
}
